// GCN3layer_5995774345733
// MI455X (gfx1250) — matrix operand order repaired and checked by reading, not compiled, not run on hardware
//
#include <hip/hip_runtime.h>
#include <math.h>

typedef __bf16 bf16;
typedef __attribute__((ext_vector_type(16))) __bf16 v16bf;
typedef __attribute__((ext_vector_type(8)))  __bf16 bf16x8;
typedef __attribute__((ext_vector_type(8)))  float  v8f;
typedef __attribute__((ext_vector_type(4)))  int    v4i;

typedef __attribute__((address_space(1))) v4i* v4i_gptr;
typedef __attribute__((address_space(3))) v4i* v4i_lptr;

#define NFEAT  512
#define NHID   128
#define NCLASS 64
#define LDS_PAD 8      // bf16 elements of row padding -> 1040B pitch, conflict-spread

#if defined(__has_builtin)
#  if __has_builtin(__builtin_amdgcn_global_load_async_to_lds_b128)
#    define HAVE_ASYNC_LDS 1
#  else
#    define HAVE_ASYNC_LDS 0
#  endif
#else
#  define HAVE_ASYNC_LDS 0
#endif

static __device__ __forceinline__ void atomAddF32(float* p, float v) {
    __hip_atomic_fetch_add(p, v, __ATOMIC_RELAXED, __HIP_MEMORY_SCOPE_AGENT);
}

static __device__ __forceinline__ bf16 f2bf(float f) { return (bf16)f; }

// ---------------- utility kernels ----------------

__global__ void k_fill(float* __restrict__ p, float v, int n) {
    int i = blockIdx.x * blockDim.x + threadIdx.x;
    if (i < n) p[i] = v;
}

__global__ void k_deg(const int* __restrict__ dst, int E, float* __restrict__ deg) {
    int e = blockIdx.x * blockDim.x + threadIdx.x;
    if (e < E) atomAddF32(&deg[dst[e]], 1.0f);
}

__global__ void k_rsqrt_inplace(float* __restrict__ d, int n) {
    int i = blockIdx.x * blockDim.x + threadIdx.x;
    if (i < n) d[i] = rsqrtf(d[i]);   // deg >= 1 (self loop) so always valid
}

__global__ void k_cvt_bf16(const float* __restrict__ in, bf16* __restrict__ out, int n) {
    int i = blockIdx.x * blockDim.x + threadIdx.x;
    if (i < n) out[i] = f2bf(in[i]);
}

// W: K x N row-major f32  ->  Wt: N x K row-major bf16
__global__ void k_transpose_cvt(const float* __restrict__ W, bf16* __restrict__ Wt,
                                int K, int N) {
    int i = blockIdx.x * blockDim.x + threadIdx.x;
    if (i < K * N) {
        int k = i / N, n = i % N;
        Wt[(long)n * K + k] = f2bf(W[i]);
    }
}

// ---------------- WMMA GEMM: C[MxN] = A[MxK] * Bt[NxK]^T (bf16 in, f32 out) ----
// Block = 256 threads = 8 waves. Each block: 8 consecutive 16-row tiles x one
// 64-col strip. Bt strip staged in LDS once (async DMA when available, shared
// by all waves). A fragments double-buffered from global; B fragments
// double-buffered from LDS so every ds_load pair is issued one WMMA ahead of
// its consumer. M%16==0, N%64==0, K%32==0.
__global__ void __launch_bounds__(256)
k_gemm_wmma_bf16(const bf16* __restrict__ A, const bf16* __restrict__ Bt,
                 float* __restrict__ C, int M, int K, int N) {
    extern __shared__ bf16 sB[];               // 64 * (K + LDS_PAD) bf16

    const int strips   = N >> 6;
    const int blockRow = (int)blockIdx.x / strips;
    const int strip    = (int)blockIdx.x % strips;
    const int nb       = strip << 6;
    const int pitch    = K + LDS_PAD;

    // ---- cooperative stage of the 64 x K weight strip into LDS
    {
        const int kc = K >> 3;                 // 8-elem (16B) chunks per row
        const int nchunks = kc << 6;           // 64 rows
        for (int c = threadIdx.x; c < nchunks; c += 256) {
            int n  = c / kc;
            int k8 = (c - n * kc) << 3;
            const bf16* gsrc = Bt + ((long)(nb + n)) * K + k8;
            bf16*       ldst = &sB[n * pitch + k8];
#if HAVE_ASYNC_LDS
            __builtin_amdgcn_global_load_async_to_lds_b128(
                (v4i_gptr)(bf16*)gsrc, (v4i_lptr)ldst, 0, 0);
#else
            *(bf16x8*)ldst = *(const bf16x8*)gsrc;
#endif
        }
    }
#if HAVE_ASYNC_LDS
    asm volatile("s_wait_asynccnt 0x0" ::: "memory");
#endif
    __syncthreads();

    const int lane    = threadIdx.x & 31;
    const int waveId  = threadIdx.x >> 5;
    const int rowTile = blockRow * 8 + waveId;
    if (rowTile >= (M >> 4)) return;           // after barrier: safe

    const int hi = lane >> 4;
    const int lo = lane & 15;
    const int m  = (rowTile << 4) + lo;
    const bf16* __restrict__ arow = A + (long)m * K;

    v8f acc[4];
    #pragma unroll
    for (int t = 0; t < 4; ++t) acc[t] = (v8f){0.f,0.f,0.f,0.f,0.f,0.f,0.f,0.f};

    // A fragment loader: ISA 16-bit A layout
    //   elems 0..7  -> K = kk + 8*hi + {0..7};  elems 8..15 -> K = kk+16+8*hi+{0..7}
    auto loadA = [&](int kk) -> v16bf {
        bf16x8 c0 = *(const bf16x8*)(arow + kk + 8 * hi);
        bf16x8 c1 = *(const bf16x8*)(arow + kk + 16 + 8 * hi);
        v16bf a;
        #pragma unroll
        for (int i = 0; i < 8; ++i) { a[i] = c0[i]; a[8 + i] = c1[i]; }
        return a;
    };
    // B fragment loader from LDS: lane's column (tile t), K = kk+8*hi+{0..7}, kk+16+8*hi+{0..7}
    auto loadB = [&](int t, int kk) -> v16bf {
        int nl = (t << 4) + lo;
        const bf16* bp = &sB[nl * pitch + kk + 8 * hi];
        bf16x8 b0 = *(const bf16x8*)(bp);
        bf16x8 b1 = *(const bf16x8*)(bp + 16);
        v16bf b;
        #pragma unroll
        for (int i = 0; i < 8; ++i) { b[i] = b0[i]; b[8 + i] = b1[i]; }
        return b;
    };

    v16bf a_cur = loadA(0);
    v16bf b_cur = loadB(0, 0);
    for (int kk = 0; kk < K; kk += 32) {
        int kn = (kk + 32 < K) ? (kk + 32) : kk;   // next K-step (clamped)
        v16bf a_nxt = loadA(kn);                   // overlap global latency

        #pragma unroll
        for (int t = 0; t < 4; ++t) {
            // issue next B fragment's ds_loads before consuming current one
            v16bf b_nxt = (t < 3) ? loadB(t + 1, kk) : loadB(0, kn);
            acc[t] = __builtin_amdgcn_wmma_f32_16x16x32_bf16(
                false, a_cur, false, b_cur, (short)0, acc[t], false, false);
            b_cur = b_nxt;
        }
        a_cur = a_nxt;
    }

    // Writeback: acc[t] elem r -> row = rowTile*16 + r + 8*hi, col = nb + t*16 + lo
    #pragma unroll
    for (int t = 0; t < 4; ++t) {
        int n = nb + (t << 4) + lo;
        #pragma unroll
        for (int r = 0; r < 8; ++r) {
            int row = (rowTile << 4) + r + 8 * hi;
            C[(long)row * N + n] = acc[t][r];
        }
    }
}

// ---------------- edge aggregation: out[dst] += norm * h[src] -----------------
// one wave per edge; lane strides over features
__global__ void k_agg_edges(const int* __restrict__ src, const int* __restrict__ dst,
                            const float* __restrict__ dinv,
                            const float* __restrict__ h, float* __restrict__ out,
                            int E, int F) {
    int gid  = blockIdx.x * blockDim.x + threadIdx.x;
    int wave = gid >> 5, lane = gid & 31;
    if (wave >= E) return;
    int s = src[wave], d = dst[wave];
    float norm = dinv[s] * dinv[d];
    const float* hr = h + (long)s * F;
    float* orow = out + (long)d * F;
    for (int f = lane; f < F; f += 32)
        atomAddF32(&orow[f], norm * hr[f]);
}

// self-loop contribution + bias + optional relu
__global__ void k_epilogue(const float* __restrict__ h, const float* __restrict__ dinv,
                           const float* __restrict__ bias, float* __restrict__ out,
                           int Nn, int F, int doRelu) {
    int i = blockIdx.x * blockDim.x + threadIdx.x;
    if (i >= Nn * F) return;
    int node = i / F, f = i % F;
    float di = dinv[node];
    float v = out[i] + h[i] * di * di + bias[f];
    out[i] = doRelu ? fmaxf(v, 0.0f) : v;
}

// ---------------- log_softmax over 64 classes, one wave per row ---------------
__global__ void k_logsoftmax64(const float* __restrict__ in, float* __restrict__ out,
                               int Nn) {
    int gid  = blockIdx.x * blockDim.x + threadIdx.x;
    int wave = gid >> 5, lane = gid & 31;
    if (wave >= Nn) return;
    const float* r = in + (long)wave * 64;
    float v0 = r[lane], v1 = r[lane + 32];
    float mx = fmaxf(v0, v1);
    #pragma unroll
    for (int o = 16; o > 0; o >>= 1) mx = fmaxf(mx, __shfl_xor(mx, o, 32));
    float s = __expf(v0 - mx) + __expf(v1 - mx);
    #pragma unroll
    for (int o = 16; o > 0; o >>= 1) s += __shfl_xor(s, o, 32);
    float lse = mx + __logf(s);
    float* w = out + (long)wave * 64;
    w[lane]      = v0 - lse;
    w[lane + 32] = v1 - lse;
}

// ---------------- host launcher ----------------

extern "C" void kernel_launch(void* const* d_in, const int* in_sizes, int n_in,
                              void* d_out, int out_size, void* d_ws, size_t ws_size,
                              hipStream_t stream) {
    const float* x  = (const float*)d_in[0];
    const int*   ei = (const int*)d_in[1];
    const float* W1 = (const float*)d_in[2];
    const float* b1 = (const float*)d_in[3];
    const float* W2 = (const float*)d_in[4];
    const float* b2 = (const float*)d_in[5];
    const float* W3 = (const float*)d_in[6];
    const float* b3 = (const float*)d_in[7];

    const int Nn = in_sizes[0] / NFEAT;   // 50000
    const int E  = in_sizes[1] / 2;       // 800000
    const int* src = ei;
    const int* dst = ei + E;

    // workspace carving, 256B aligned
    char* ws = (char*)d_ws;
    size_t off = 0;
    auto carve = [&](size_t bytes) -> char* {
        char* p = ws + off;
        off += (bytes + 255) & ~(size_t)255;
        return p;
    };
    float* dinv = (float*)carve((size_t)Nn * 4);
    bf16*  xb   = (bf16*) carve((size_t)Nn * NFEAT * 2);
    bf16*  w1t  = (bf16*) carve((size_t)NFEAT * NHID * 2);
    bf16*  w2t  = (bf16*) carve((size_t)NHID * NHID * 2);
    bf16*  w3t  = (bf16*) carve((size_t)NHID * NCLASS * 2);
    float* hbuf = (float*)carve((size_t)Nn * NHID * 4);
    float* abuf = (float*)carve((size_t)Nn * NHID * 4);
    bf16*  hb   = (bf16*) carve((size_t)Nn * NHID * 2);

    const int B = 256;
    auto blocks = [&](long n) -> int { return (int)((n + B - 1) / B); };

    // ---- normalization: deg (with self-loop) -> dinv = rsqrt(deg)
    k_fill<<<blocks(Nn), B, 0, stream>>>(dinv, 1.0f, Nn);
    k_deg<<<blocks(E), B, 0, stream>>>(dst, E, dinv);
    k_rsqrt_inplace<<<blocks(Nn), B, 0, stream>>>(dinv, Nn);

    // ---- precision conversions
    const int nx = Nn * NFEAT;
    k_cvt_bf16<<<blocks(nx), B, 0, stream>>>(x, xb, nx);
    k_transpose_cvt<<<blocks(NFEAT * NHID), B, 0, stream>>>(W1, w1t, NFEAT, NHID);
    k_transpose_cvt<<<blocks(NHID * NHID), B, 0, stream>>>(W2, w2t, NHID, NHID);
    k_transpose_cvt<<<blocks(NHID * NCLASS), B, 0, stream>>>(W3, w3t, NHID, NCLASS);

    auto gemm = [&](const bf16* Ab, const bf16* Bt, float* C, int M, int K, int N) {
        int strips = N / 64;
        int nBlockRows = (M / 16 + 7) / 8;             // 8 row-tiles per block
        size_t ldsBytes = (size_t)64 * (K + LDS_PAD) * sizeof(bf16);
        k_gemm_wmma_bf16<<<nBlockRows * strips, B, ldsBytes, stream>>>(Ab, Bt, C, M, K, N);
    };
    auto agg = [&](const float* h, const float* bias, float* out, int F, int relu) {
        int n = Nn * F;
        k_fill<<<blocks(n), B, 0, stream>>>(out, 0.0f, n);
        k_agg_edges<<<blocks((long)E * 32), B, 0, stream>>>(src, dst, dinv, h, out, E, F);
        k_epilogue<<<blocks(n), B, 0, stream>>>(h, dinv, bias, out, Nn, F, relu);
    };

    const int nh = Nn * NHID;

    // ---- layer 1: h = relu(agg(x @ W1) + b1)
    gemm(xb, w1t, hbuf, Nn, NFEAT, NHID);
    agg(hbuf, b1, abuf, NHID, 1);
    k_cvt_bf16<<<blocks(nh), B, 0, stream>>>(abuf, hb, nh);

    // ---- layer 2
    gemm(hb, w2t, hbuf, Nn, NHID, NHID);
    agg(hbuf, b2, abuf, NHID, 1);
    k_cvt_bf16<<<blocks(nh), B, 0, stream>>>(abuf, hb, nh);

    // ---- layer 3 (N=64, no relu) + log_softmax
    gemm(hb, w3t, hbuf, Nn, NHID, NCLASS);
    agg(hbuf, b3, abuf, NCLASS, 0);
    k_logsoftmax64<<<blocks((long)Nn * 32), B, 0, stream>>>(abuf, (float*)d_out, Nn);
}
